// InputMPNN_45234595562118
// MI455X (gfx1250) — compile-verified, matrix operand order checked by reading, not run on hardware
//
#include <hip/hip_runtime.h>
#include <hip/hip_bf16.h>

// ---------------- types / helpers ----------------
typedef __attribute__((ext_vector_type(16))) __bf16 v16bf;
typedef __attribute__((ext_vector_type(8)))  float  v8f;

union Frag {
    v16bf v;
    unsigned int u[8];
};

__device__ __forceinline__ unsigned short bf_bits(float f) {
    union { __bf16 h; unsigned short u; } t;
    t.h = (__bf16)f;
    return t.u;
}
__device__ __forceinline__ unsigned int pack2(float a, float b) {
    union { __bf16 h[2]; unsigned int u; } t;
    t.h[0] = (__bf16)a;
    t.h[1] = (__bf16)b;
    return t.u;
}
__device__ __forceinline__ v8f wmma_bf16(const Frag& a, const Frag& b, v8f c) {
    return __builtin_amdgcn_wmma_f32_16x16x32_bf16(false, a.v, false, b.v,
                                                   (short)0, c, false, false);
}
__device__ __forceinline__ Frag ld_frag_lds(const __bf16* p) {
    uint4 lo = *(const uint4*)(p);
    uint4 hi = *(const uint4*)(p + 16);
    Frag f;
    f.u[0] = lo.x; f.u[1] = lo.y; f.u[2] = lo.z; f.u[3] = lo.w;
    f.u[4] = hi.x; f.u[5] = hi.y; f.u[6] = hi.z; f.u[7] = hi.w;
    return f;
}
__device__ __forceinline__ Frag ld_frag_glb(const unsigned short* p) {
    uint4 lo = *(const uint4*)(p);
    uint4 hi = *(const uint4*)(p + 16);
    Frag f;
    f.u[0] = lo.x; f.u[1] = lo.y; f.u[2] = lo.z; f.u[3] = lo.w;
    f.u[4] = hi.x; f.u[5] = hi.y; f.u[6] = hi.z; f.u[7] = hi.w;
    return f;
}

static constexpr int AN   = 512;   // atoms
static constexpr int CIN  = 64;
static constexpr int FB_S = 520;   // padded LDS stride (bf16) -> conflict-free b128 tiles
static constexpr int H2_S = 260;   // padded LDS stride for h2 tile (floats)

// ======================================================================
// Kernel 0: convert w1 (256x128) and w2 (128x256) f32 -> bf16 in ws
// ======================================================================
__global__ void k_cvt_w(const float* __restrict__ w1, const float* __restrict__ w2,
                        unsigned short* __restrict__ w1b, unsigned short* __restrict__ w2b) {
    int idx = blockIdx.x * 256 + threadIdx.x;
    if (idx < 32768) {
        w1b[idx] = bf_bits(w1[idx]);
        w2b[idx] = bf_bits(w2[idx]);
    }
}

// ======================================================================
// Kernel 1: message passing as  T[k,c] = sum_x G^T[k,x] * F[x,c]  (WMMA GEMM,
//           A = G^T and B = features, both streamed from LDS), then
//           mp[c] = sum_k w[c,k] * T[k,c].
//   - G^T (32 x 512) precomputed cooperatively per atom into LDS (bf16),
//     cutoff folded in; row k=0 (sin(0)==0 for this basis) carries cut[x]
//     so the rad_b*cut*feat bias term rides the same GEMM with weight rad_b.
//   - One WG = 1 batch, 8 target atoms; wave w = channel tile [16w,16w+16).
// ======================================================================
__global__ void k_edge_mp(const float* __restrict__ features,
                          const unsigned char* __restrict__ edge_mask,
                          const float* __restrict__ norms,
                          const float* __restrict__ scales,
                          const float* __restrict__ phases,
                          const float* __restrict__ rad_w,
                          const float* __restrict__ rad_b,
                          const float* __restrict__ soft_rad,
                          const float* __restrict__ soft_width,
                          unsigned short* __restrict__ hbuf) {
    extern __shared__ __align__(16) unsigned short sh[];
    __bf16* Ftb = (__bf16*)sh;                 // [64][FB_S]  features^T, bf16
    __bf16* Gbf = (__bf16*)sh + 64 * FB_S;     // [32][FB_S]  G^T for current atom

    const int g    = blockIdx.x;               // 0..255
    const int b    = g >> 6;
    const int i0   = (g & 63) << 3;            // 8 atoms per WG
    const int tid  = threadIdx.x;
    const int wave = tid >> 5;                 // channel tile
    const int lane = tid & 31;
    const int hl   = lane >> 4;
    const int lr   = lane & 15;

    // ---- stage features transposed as bf16 ----
    for (int idx = tid; idx < AN * CIN; idx += 128) {
        int x = idx >> 6, c = idx & 63;
        Ftb[c * FB_S + x] = (__bf16)features[((size_t)b * AN + x) * CIN + c];
    }
    __syncthreads();

    // ---- per-lane combine weights: w[k] = (k==0 ? rad_b[c] : rad_w[c][k]) ----
    float wA[8], wB[8];
    {
        int c = wave * 16 + lr;
        #pragma unroll
        for (int v = 0; v < 8; ++v) {
            int kA = v + hl * 8;
            int kB = 16 + hl * 8 + v;
            wA[v] = (kA == 0) ? rad_b[c] : rad_w[c * 32 + kA];
            wB[v] = rad_w[c * 32 + kB];
        }
    }

    const float srad = soft_rad[0], swid = soft_width[0];
    float sc[8], phs[8];
    #pragma unroll
    for (int t = 0; t < 8; ++t) { sc[t] = scales[t]; phs[t] = phases[t]; }
    const float TWO_PI = 6.283185307179586f;
    const int x0 = tid * 4;                    // 4 neighbor columns per thread

    // loop-invariant LDS bases (immediate offsets inside the K loop)
    const __bf16* bbase = Ftb + (wave * 16 + lr) * FB_S + hl * 8;   // B: features
    const __bf16* abase = Gbf + lr * FB_S + hl * 8;                 // A: G^T rows 0..15

    #pragma unroll 1
    for (int ii = 0; ii < 8; ++ii) {
        const int i = i0 + ii;
        if (ii) __syncthreads();               // previous GEMM done reading Gbf

        // ---- cooperative precompute of G^T rows k=4t+p (each sin computed once) ----
        {
            const float*         nrow = norms     + ((size_t)b * AN + i) * AN;
            const unsigned char* erow = edge_mask + ((size_t)b * AN + i) * AN;
            float4 rv = *(const float4*)(nrow + x0);
            float r4[4] = {rv.x, rv.y, rv.z, rv.w};
            float fac[4], cutv[4], ipw[4][4];
            #pragma unroll
            for (int j = 0; j < 4; ++j) {
                bool  edm  = erow[x0 + j] != 0;
                float r    = r4[j];
                bool  em   = edm && (r > 0.f);
                float soft = 1.f / (1.f + __expf(-(srad - r) / swid));
                cutv[j] = (edm && (r < 1.73f)) ? soft : 0.f;   // hard * soft cutoff
                fac[j]  = em ? cutv[j] : 0.f;                  // + valid-edge gate
                float inv = 1.f / (em ? r : 1.f);
                ipw[0][j] = 1.f; ipw[1][j] = inv;
                ipw[2][j] = inv * inv; ipw[3][j] = inv * inv * inv;
            }
            #pragma unroll
            for (int t = 0; t < 8; ++t) {
                float s4[4];
                #pragma unroll
                for (int j = 0; j < 4; ++j)
                    s4[j] = __sinf(TWO_PI * sc[t] * r4[j] + phs[t]);
                #pragma unroll
                for (int p = 0; p < 4; ++p) {
                    int k = 4 * t + p;
                    float v0, v1, v2, v3;
                    if (k == 0) {                 // dead basis row carries cut[x]
                        v0 = cutv[0]; v1 = cutv[1]; v2 = cutv[2]; v3 = cutv[3];
                    } else {
                        v0 = fac[0] * s4[0] * ipw[p][0];
                        v1 = fac[1] * s4[1] * ipw[p][1];
                        v2 = fac[2] * s4[2] * ipw[p][2];
                        v3 = fac[3] * s4[3] * ipw[p][3];
                    }
                    uint2 pk;
                    pk.x = pack2(v0, v1);
                    pk.y = pack2(v2, v3);
                    *(uint2*)(Gbf + k * FB_S + x0) = pk;
                }
            }
        }
        __syncthreads();

        // ---- T = G^T @ F : 16 K-steps, 2 M-tiles, all operands from LDS ----
        v8f acc0 = {}, acc1 = {};
        #pragma unroll
        for (int ks = 0; ks < 16; ++ks) {
            Frag bf = ld_frag_lds(bbase + ks * 32);               // features chunk
            Frag a0 = ld_frag_lds(abase + ks * 32);               // G^T rows 0..15
            Frag a1 = ld_frag_lds(abase + 16 * FB_S + ks * 32);   // G^T rows 16..31
            acc0 = wmma_bf16(a0, bf, acc0);
            acc1 = wmma_bf16(a1, bf, acc1);
        }

        // ---- epilogue: mp[c] = sum_k w[k] * T[k][c] ----
        float mp = 0.f;
        #pragma unroll
        for (int v = 0; v < 8; ++v) mp += wA[v] * acc0[v] + wB[v] * acc1[v];
        mp += __shfl_xor(mp, 16, 32);          // halves hold disjoint k rows
        if (hl == 0) {
            int c = wave * 16 + lr;
            size_t base = ((size_t)b * AN + i) * 128;
            hbuf[base + c]      = bf_bits(mp);
            hbuf[base + 64 + c] = ((const unsigned short*)Ftb)[c * FB_S + i];
        }
    }
}

// ======================================================================
// Kernel 2: per-atom MLP as a WMMA GEMM (16 atom rows per wave).
// ======================================================================
__global__ void k_mlp(const unsigned short* __restrict__ hbuf,
                      const unsigned short* __restrict__ w1b,
                      const float* __restrict__ b1,
                      const unsigned short* __restrict__ w2b,
                      const float* __restrict__ b2,
                      const unsigned char* __restrict__ atom_mask,
                      float* __restrict__ out) {
    extern __shared__ float smem[];
    const int tid  = threadIdx.x;
    const int wave = tid >> 5;
    const int lane = tid & 31;
    const int hl   = lane >> 4;
    const int lr   = lane & 15;
    float* h2 = smem + wave * (16 * H2_S);

    const int mt  = blockIdx.x * 4 + wave;
    const int m0  = mt * 16;
    const int row = m0 + lr;

    Frag a1[4];
    #pragma unroll
    for (int ks = 0; ks < 4; ++ks)
        a1[ks] = ld_frag_glb(hbuf + (size_t)row * 128 + ks * 32 + hl * 8);

    for (int nt = 0; nt < 16; ++nt) {
        v8f acc = {};
        #pragma unroll
        for (int ks = 0; ks < 4; ++ks) {
            Frag bf = ld_frag_glb(w1b + (size_t)(nt * 16 + lr) * 128 + ks * 32 + hl * 8);
            acc = wmma_bf16(a1[ks], bf, acc);
        }
        const int c = nt * 16 + lr;
        const float bias = b1[c];
        #pragma unroll
        for (int v = 0; v < 8; ++v) {
            float x = acc[v] + bias;
            x = x > 0.f ? x : 0.01f * x;
            h2[(v + hl * 8) * H2_S + c] = x;
        }
    }
    asm volatile("s_wait_dscnt 0" ::: "memory");

    Frag a2[8];
    #pragma unroll
    for (int ks = 0; ks < 8; ++ks) {
        const float* p = h2 + lr * H2_S + ks * 32 + hl * 8;
        float4 q0 = *(const float4*)(p);
        float4 q1 = *(const float4*)(p + 4);
        float4 q2 = *(const float4*)(p + 16);
        float4 q3 = *(const float4*)(p + 20);
        a2[ks].u[0] = pack2(q0.x, q0.y); a2[ks].u[1] = pack2(q0.z, q0.w);
        a2[ks].u[2] = pack2(q1.x, q1.y); a2[ks].u[3] = pack2(q1.z, q1.w);
        a2[ks].u[4] = pack2(q2.x, q2.y); a2[ks].u[5] = pack2(q2.z, q2.w);
        a2[ks].u[6] = pack2(q3.x, q3.y); a2[ks].u[7] = pack2(q3.z, q3.w);
    }

    for (int nt = 0; nt < 8; ++nt) {
        v8f acc = {};
        #pragma unroll
        for (int ks = 0; ks < 8; ++ks) {
            Frag bf = ld_frag_glb(w2b + (size_t)(nt * 16 + lr) * 256 + ks * 32 + hl * 8);
            acc = wmma_bf16(a2[ks], bf, acc);
        }
        const int c    = nt * 16 + lr;
        const float bs = b2[c];
        const int ch   = c & 63;
        const int part = c >> 6;
        #pragma unroll
        for (int v = 0; v < 8; ++v) {
            const int n = m0 + v + hl * 8;
            float val = acc[v] + bs;
            if (!atom_mask[n]) val = 0.f;
            out[((size_t)n * 64 + ch) * 2 + part] = val;
        }
    }
}

// ======================================================================
extern "C" void kernel_launch(void* const* d_in, const int* in_sizes, int n_in,
                              void* d_out, int out_size, void* d_ws, size_t ws_size,
                              hipStream_t stream) {
    const float*         features   = (const float*)d_in[0];
    const unsigned char* atom_mask  = (const unsigned char*)d_in[1];
    // d_in[2] = edge_features (unused by the module)
    const unsigned char* edge_mask  = (const unsigned char*)d_in[3];
    const float*         norms      = (const float*)d_in[4];
    const float*         scales     = (const float*)d_in[5];
    const float*         phases     = (const float*)d_in[6];
    const float*         rad_w      = (const float*)d_in[7];
    const float*         rad_b      = (const float*)d_in[8];
    const float*         soft_rad   = (const float*)d_in[9];
    const float*         soft_width = (const float*)d_in[10];
    const float*         w1         = (const float*)d_in[11];
    const float*         b1         = (const float*)d_in[12];
    const float*         w2         = (const float*)d_in[13];
    const float*         b2         = (const float*)d_in[14];
    float*               out        = (float*)d_out;

    // workspace: hbuf (2048x128 bf16 = 512KB) | w1b (64KB) | w2b (64KB)
    unsigned short* hbuf = (unsigned short*)d_ws;
    unsigned short* w1b  = (unsigned short*)((char*)d_ws + 512 * 1024);
    unsigned short* w2b  = (unsigned short*)((char*)d_ws + 576 * 1024);

    k_cvt_w<<<128, 256, 0, stream>>>(w1, w2, w1b, w2b);

    const size_t smem1 = (size_t)(64 + 32) * FB_S * sizeof(unsigned short);  // 99840 B
    k_edge_mp<<<256, 128, smem1, stream>>>(features, edge_mask, norms, scales, phases,
                                           rad_w, rad_b, soft_rad, soft_width, hbuf);

    const size_t smem2 = (size_t)4 * 16 * H2_S * sizeof(float);              // 66560 B
    k_mlp<<<32, 128, smem2, stream>>>(hbuf, w1b, b1, w2b, b2, atom_mask, out);
}
